// BiLSTMCRFForTokenClassification_87875030876970
// MI455X (gfx1250) — compile-verified, matrix operand order checked by reading, not run on hardware
//
#include <hip/hip_runtime.h>
#include <hip/hip_bf16.h>
#include <math.h>

// BiLSTM-CRF for MI455X (gfx1250, wave32).
// GEMMs use V_WMMA_F32_16X16X32_BF16 (bf16 operands, f32 accum).
// LSTM scan = persistent kernel; Whh tile pinned in VGPRs; h_prev staged into
// LDS each step via GLOBAL_LOAD_ASYNC_TO_LDS_B128 (ASYNCcnt); grid barrier/step.

typedef __attribute__((ext_vector_type(16))) __bf16 v16bf;
typedef __attribute__((ext_vector_type(8)))  float  v8f;

#define CV 50257   // vocab
#define CE 256     // embed
#define CH 512     // hidden
#define CT 9       // tags
#define CB 32      // batch
#define CS 512     // seq
#define CG4 2048   // 4*H

__device__ __forceinline__ float sigmf(float x) { return 1.0f / (1.0f + expf(-x)); }

// ---------------------------------------------------------------- converts
__global__ void f32_to_bf16_kernel(const float* __restrict__ in,
                                   __bf16* __restrict__ out, int n) {
  int i = blockIdx.x * blockDim.x + threadIdx.x;
  if (i < n) out[i] = (__bf16)in[i];
}

// Wc [T, 2H] f32 -> padded bf16 [16][1024] (rows 9..15 = 0)
__global__ void wc_pad_kernel(const float* __restrict__ Wc, __bf16* __restrict__ out) {
  int i = blockIdx.x * blockDim.x + threadIdx.x;  // 16*1024
  if (i >= 16 * 1024) return;
  int row = i >> 10, col = i & 1023;
  out[i] = (row < CT) ? (__bf16)Wc[row * (2 * CH) + col] : (__bf16)0.0f;
}

// zero h(t=0) state + reset grid-barrier counter
__global__ void init_scan_kernel(__bf16* __restrict__ h0, int* __restrict__ bar) {
  int i = blockIdx.x * blockDim.x + threadIdx.x;
  if (i < CB * CH) h0[i] = (__bf16)0.0f;
  if (i == 0) *bar = 0;
}

// ------------------------------------------------- embedding gather + x@Wih.T
// xproj[s][b][n] = sum_k emb[ids[b][s]][k] * Wih[n][k] + bih[n] + bhh[n]
// grid: (1024, 16), block 256 (8 waves). wave -> one 16x16 N-tile.
__global__ __launch_bounds__(256) void embed_proj_kernel(
    const int* __restrict__ ids,        // [B][S]
    const __bf16* __restrict__ embb,    // [V][E] bf16
    const __bf16* __restrict__ Wih,     // [4H][E] bf16
    const float* __restrict__ bih, const float* __restrict__ bhh,
    __bf16* __restrict__ xproj) {       // [S][B][4H] bf16
  int wave = threadIdx.x >> 5, lane = threadIdx.x & 31;
  int mt = blockIdx.x;                       // 0..1023  (rows r = s*B + b)
  int n0 = (blockIdx.y * 8 + wave) * 16;     // 0..2032
  int r  = mt * 16 + (lane & 15);
  int s  = r >> 5, b = r & 31;               // B = 32
  int id = ids[b * CS + s];
  int kh = (lane >> 4) * 16;
  const __bf16* Arow = embb + (size_t)id * CE + kh;
  const __bf16* Brow = Wih + (size_t)(n0 + (lane & 15)) * CE + kh;
  v8f acc = {};
#pragma unroll
  for (int k = 0; k < CE; k += 32) {
    v16bf av = *(const v16bf*)(Arow + k);
    v16bf bv = *(const v16bf*)(Brow + k);
    acc = __builtin_amdgcn_wmma_f32_16x16x32_bf16(false, av, false, bv,
                                                  (short)0, acc, false, false);
  }
  int col = n0 + (lane & 15);
  float bias = bih[col] + bhh[col];
  int rbase = mt * 16 + ((lane >> 4) * 8);
#pragma unroll
  for (int v = 0; v < 8; v++) {
    int rr = rbase + v;
    int s2 = rr >> 5, b2 = rr & 31;
    xproj[(size_t)s2 * CB * CG4 + (size_t)b2 * CG4 + col] = (__bf16)(acc[v] + bias);
  }
}

// ---------------------------------------------------- persistent LSTM scan
// 32 blocks (H/16 col tiles) x 256 threads (8 waves: 4 gates x 2 M-tiles).
// Whh tile lives in VGPRs the whole scan; c state in registers; h_prev staged
// into LDS via async-to-LDS each step; h ping-pongs through global memory.
__global__ __launch_bounds__(256) void lstm_scan_kernel(
    const __bf16* __restrict__ xproj,   // [S][B][4H]
    const __bf16* __restrict__ Whh,     // [4H][H] bf16
    __bf16* __restrict__ hping,         // [2][B][H] (buffer 0 pre-zeroed)
    __bf16* __restrict__ hs,            // [B][S][H]
    int* __restrict__ bar, int reverse) {
  __shared__ float  gsm[4][CB][16];     // 8 KB gate staging
  __shared__ __bf16 hsm[CB * CH];       // 32 KB h_prev stage
  int wave = threadIdx.x >> 5, lane = threadIdx.x & 31;
  int gate = wave >> 1;
  int m0   = (wave & 1) * 16;
  int j0   = blockIdx.x * 16;
  int kh   = (lane >> 4) * 16;
  int colL = lane & 15;
  int rowb = m0 + ((lane >> 4) * 8);

  // pin this wave's Whh 16(col)x512(K) tile in registers: 16 x v16bf = 128 VGPRs
  v16bf bfrag[16];
  {
    const __bf16* Brow = Whh + (size_t)(gate * CH + j0 + colL) * CH + kh;
#pragma unroll
    for (int kk = 0; kk < 16; kk++) bfrag[kk] = *(const v16bf*)(Brow + kk * 32);
  }

  // per-thread cell state: elements (threadIdx.x) and (threadIdx.x+256)
  float creg[2] = {0.0f, 0.0f};

  const __bf16* hprev = hping;
  __bf16* hnext = hping + CB * CH;

  const __bf16* hrow = hsm + (size_t)(m0 + (lane & 15)) * CH + kh;  // A frags (LDS)
  unsigned int lds_base = (unsigned int)(size_t)(&hsm[0]) + threadIdx.x * 16u;

  for (int step = 0; step < CS; step++) {
    int t = reverse ? (CS - 1 - step) : step;
    const __bf16* xp = xproj + (size_t)t * CB * CG4;
    // prefetch next timestep's xproj slice while we compute (global_prefetch_b8)
    if (step + 1 < CS) {
      int tn = reverse ? (CS - 2 - step) : (step + 1);
      __builtin_prefetch(xproj + (size_t)tn * CB * CG4 + threadIdx.x * 256, 0, 0);
    }

    // ---- async copy h_prev (32 KB) global -> LDS: 256 lanes x 16 B x 8 ----
    {
      unsigned long long ga = (unsigned long long)(size_t)hprev + threadIdx.x * 16ull;
      unsigned int la = lds_base;
      asm volatile("global_load_async_to_lds_b128 %0, %1, off"               :: "v"(la), "v"(ga) : "memory");
      asm volatile("global_load_async_to_lds_b128 %0, %1, off offset:4096"   :: "v"(la), "v"(ga) : "memory");
      asm volatile("global_load_async_to_lds_b128 %0, %1, off offset:8192"   :: "v"(la), "v"(ga) : "memory");
      asm volatile("global_load_async_to_lds_b128 %0, %1, off offset:12288"  :: "v"(la), "v"(ga) : "memory");
      asm volatile("global_load_async_to_lds_b128 %0, %1, off offset:16384"  :: "v"(la), "v"(ga) : "memory");
      asm volatile("global_load_async_to_lds_b128 %0, %1, off offset:20480"  :: "v"(la), "v"(ga) : "memory");
      asm volatile("global_load_async_to_lds_b128 %0, %1, off offset:24576"  :: "v"(la), "v"(ga) : "memory");
      asm volatile("global_load_async_to_lds_b128 %0, %1, off offset:28672"  :: "v"(la), "v"(ga) : "memory");
      asm volatile("s_wait_asynccnt 0x0" ::: "memory");
    }
    __syncthreads();  // all lanes' copies complete -> hsm valid for everyone

    v8f acc = {};
#pragma unroll
    for (int kk = 0; kk < 16; kk++) {
      v16bf av = *(const v16bf*)(hrow + kk * 32);   // ds_load from staged h
      acc = __builtin_amdgcn_wmma_f32_16x16x32_bf16(false, av, false, bfrag[kk],
                                                    (short)0, acc, false, false);
    }
#pragma unroll
    for (int v = 0; v < 8; v++) gsm[gate][rowb + v][colL] = acc[v];
    __syncthreads();

#pragma unroll
    for (int u = 0; u < 2; u++) {
      int e = threadIdx.x + u * 256;          // 512 elems = B x 16 cols
      int b = e >> 4, jl = e & 15, j = j0 + jl;
      float gi = gsm[0][b][jl] + (float)xp[b * CG4 + 0 * CH + j];
      float gf = gsm[1][b][jl] + (float)xp[b * CG4 + 1 * CH + j];
      float gg = gsm[2][b][jl] + (float)xp[b * CG4 + 2 * CH + j];
      float go = gsm[3][b][jl] + (float)xp[b * CG4 + 3 * CH + j];
      float c = sigmf(gf) * creg[u] + sigmf(gi) * tanhf(gg);
      float h = sigmf(go) * tanhf(c);
      creg[u] = c;
      hnext[b * CH + j] = (__bf16)h;
      hs[(size_t)b * CS * CH + (size_t)t * CH + j] = (__bf16)h;
    }

    // ---- grid barrier (monotonic counter; also fences hsm/gsm reuse) ----
    __threadfence();
    if (threadIdx.x == 0) {
      __hip_atomic_fetch_add(bar, 1, __ATOMIC_RELEASE, __HIP_MEMORY_SCOPE_AGENT);
      int target = (step + 1) * (int)gridDim.x;
      while (__hip_atomic_load(bar, __ATOMIC_ACQUIRE, __HIP_MEMORY_SCOPE_AGENT) < target) {
        __builtin_amdgcn_s_sleep(1);
      }
    }
    __syncthreads();
    __threadfence();

    // swap h ping-pong
    const __bf16* tmp = hprev;
    hprev = hnext;
    hnext = (__bf16*)tmp;
  }
}

// ------------------------------------------------- classifier: out @ Wc.T + bc
// A = [h_f | h_b] rows r = b*S+s (M=16384, K=1024), B = padded Wc (N=16).
__global__ __launch_bounds__(256) void classifier_kernel(
    const __bf16* __restrict__ hsf, const __bf16* __restrict__ hsb,
    const __bf16* __restrict__ Wcp,   // [16][1024] bf16 (padded)
    const float* __restrict__ bc,
    float* __restrict__ logits) {     // [B][S][T]
  int wave = threadIdx.x >> 5, lane = threadIdx.x & 31;
  int mt = blockIdx.x * 8 + wave;     // 0..1023
  int r  = mt * 16 + (lane & 15);
  int b  = r >> 9, s = r & 511;
  int kh = (lane >> 4) * 16;
  const __bf16* Af = hsf + ((size_t)b * CS + s) * CH + kh;
  const __bf16* Ab = hsb + ((size_t)b * CS + s) * CH + kh;
  const __bf16* Br = Wcp + (size_t)(lane & 15) * 1024 + kh;
  v8f acc = {};
#pragma unroll 4
  for (int k = 0; k < CH; k += 32) {
    v16bf av = *(const v16bf*)(Af + k);
    v16bf bv = *(const v16bf*)(Br + k);
    acc = __builtin_amdgcn_wmma_f32_16x16x32_bf16(false, av, false, bv,
                                                  (short)0, acc, false, false);
  }
#pragma unroll 4
  for (int k = 0; k < CH; k += 32) {
    v16bf av = *(const v16bf*)(Ab + k);
    v16bf bv = *(const v16bf*)(Br + CH + k);
    acc = __builtin_amdgcn_wmma_f32_16x16x32_bf16(false, av, false, bv,
                                                  (short)0, acc, false, false);
  }
  int col = lane & 15;
  if (col < CT) {
    float bias = bc[col];
    int rbase = mt * 16 + ((lane >> 4) * 8);
#pragma unroll
    for (int v = 0; v < 8; v++) {
      int rr = rbase + v;
      int b2 = rr >> 9, s2 = rr & 511;
      logits[((size_t)b2 * CS + s2) * CT + col] = acc[v] + bias;
    }
  }
}

// --------------------------------------------------------- CRF: log-partition
__global__ __launch_bounds__(32) void crf_logz_kernel(
    const float* __restrict__ logits, const float* __restrict__ start,
    const float* __restrict__ endv, const float* __restrict__ trans,
    float* __restrict__ logZ) {
  int b = blockIdx.x, t = threadIdx.x;
  bool valid = (t < CT);
  int tc = valid ? t : 0;
  float trcol[CT];  // this lane's incoming-transition column, kept in VGPRs
#pragma unroll
  for (int f = 0; f < CT; f++) trcol[f] = trans[f * CT + tc];
  float sc = valid ? (start[tc] + logits[((size_t)b * CS + 0) * CT + tc]) : -1e30f;
  for (int s = 1; s < CS; s++) {
    float em = logits[((size_t)b * CS + s) * CT + tc];
    float vals[CT];
    float m = -1e30f;
#pragma unroll
    for (int f = 0; f < CT; f++) {
      float v = __shfl(sc, f, 32) + trcol[f];
      vals[f] = v;
      m = fmaxf(m, v);
    }
    float sum = 0.0f;
#pragma unroll
    for (int f = 0; f < CT; f++) sum += expf(vals[f] - m);
    sc = valid ? (m + logf(sum) + em) : -1e30f;
  }
  sc = valid ? (sc + endv[tc]) : -1e30f;
  float m = -1e30f;
  for (int f = 0; f < CT; f++) m = fmaxf(m, __shfl(sc, f, 32));
  float sum = 0.0f;
  for (int f = 0; f < CT; f++) sum += expf(__shfl(sc, f, 32) - m);
  if (t == 0) logZ[b] = m + logf(sum);
}

// gold path score; one thread per batch
__global__ __launch_bounds__(32) void crf_gold_kernel(
    const float* __restrict__ logits, const int* __restrict__ labels,
    const float* __restrict__ start, const float* __restrict__ endv,
    const float* __restrict__ trans, float* __restrict__ gold) {
  int b = threadIdx.x;
  const int* lab = labels + b * CS;
  float sc = start[lab[0]] + logits[((size_t)b * CS + 0) * CT + lab[0]];
  for (int s = 1; s < CS; s++)
    sc += logits[((size_t)b * CS + s) * CT + lab[s]] + trans[lab[s - 1] * CT + lab[s]];
  sc += endv[lab[CS - 1]];
  gold[b] = sc;
}

// viterbi decode; one wave per batch; decoded tags -> d_out[1..] as float
__global__ __launch_bounds__(32) void crf_viterbi_kernel(
    const float* __restrict__ logits, const float* __restrict__ start,
    const float* __restrict__ endv, const float* __restrict__ trans,
    int* __restrict__ hist, float* __restrict__ out) {
  int b = blockIdx.x, t = threadIdx.x;
  bool valid = (t < CT);
  int tc = valid ? t : 0;
  float trcol[CT];
#pragma unroll
  for (int f = 0; f < CT; f++) trcol[f] = trans[f * CT + tc];
  float sc = valid ? (start[tc] + logits[((size_t)b * CS + 0) * CT + tc]) : -1e30f;
  for (int s = 1; s < CS; s++) {
    float em = logits[((size_t)b * CS + s) * CT + tc];
    float bestv = -1e30f;
    int bestf = 0;
#pragma unroll
    for (int f = 0; f < CT; f++) {
      float v = __shfl(sc, f, 32) + trcol[f];
      if (v > bestv) { bestv = v; bestf = f; }
    }
    if (valid) hist[((size_t)b * CS + s) * CT + t] = bestf;
    sc = valid ? (bestv + em) : -1e30f;
  }
  sc = valid ? (sc + endv[tc]) : -1e30f;
  float bestv = -1e30f;
  int last = 0;
  for (int f = 0; f < CT; f++) {
    float v = __shfl(sc, f, 32);
    if (v > bestv) { bestv = v; last = f; }
  }
  if (t == 0) {
    int tag = last;
    out[1 + b * CS + (CS - 1)] = (float)tag;
    for (int s = CS - 1; s >= 1; s--) {
      tag = hist[((size_t)b * CS + s) * CT + tag];
      out[1 + b * CS + (s - 1)] = (float)tag;
    }
  }
}

__global__ __launch_bounds__(32) void loss_kernel(
    const float* __restrict__ gold, const float* __restrict__ logZ,
    float* __restrict__ out) {
  int b = threadIdx.x;
  float v = gold[b] - logZ[b];
  for (int o = 16; o > 0; o >>= 1) v += __shfl_down(v, o, 32);
  if (b == 0) out[0] = -v / (float)CB;
}

// ---------------------------------------------------------------- launcher
extern "C" void kernel_launch(void* const* d_in, const int* in_sizes, int n_in,
                              void* d_out, int out_size, void* d_ws, size_t ws_size,
                              hipStream_t stream) {
  (void)in_sizes; (void)n_in; (void)out_size; (void)ws_size;
  const int*   ids    = (const int*)d_in[0];
  const int*   labels = (const int*)d_in[2];
  const float* emb    = (const float*)d_in[3];
  const float* Wih_f  = (const float*)d_in[4];
  const float* Whh_f  = (const float*)d_in[5];
  const float* bih_f  = (const float*)d_in[6];
  const float* bhh_f  = (const float*)d_in[7];
  const float* Wih_b  = (const float*)d_in[8];
  const float* Whh_b  = (const float*)d_in[9];
  const float* bih_b  = (const float*)d_in[10];
  const float* bhh_b  = (const float*)d_in[11];
  const float* Wc     = (const float*)d_in[12];
  const float* bc     = (const float*)d_in[13];
  const float* start  = (const float*)d_in[14];
  const float* endv   = (const float*)d_in[15];
  const float* trans  = (const float*)d_in[16];
  float* out = (float*)d_out;

  char* ws = (char*)d_ws;
  size_t off = 0;
  auto alloc = [&](size_t bytes) -> void* {
    void* p = ws + off;
    off = (off + bytes + 255) & ~(size_t)255;
    return p;
  };
  __bf16* emb_bf = (__bf16*)alloc((size_t)CV * CE * 2);
  __bf16* WihF   = (__bf16*)alloc((size_t)CG4 * CE * 2);
  __bf16* WhhF   = (__bf16*)alloc((size_t)CG4 * CH * 2);
  __bf16* WihB   = (__bf16*)alloc((size_t)CG4 * CE * 2);
  __bf16* WhhB   = (__bf16*)alloc((size_t)CG4 * CH * 2);
  __bf16* Wcp    = (__bf16*)alloc((size_t)16 * 1024 * 2);
  __bf16* xproj  = (__bf16*)alloc((size_t)CS * CB * CG4 * 2);
  __bf16* hsF    = (__bf16*)alloc((size_t)CB * CS * CH * 2);
  __bf16* hsB    = (__bf16*)alloc((size_t)CB * CS * CH * 2);
  __bf16* hping  = (__bf16*)alloc((size_t)2 * CB * CH * 2);
  float* logits = (float*)alloc((size_t)CB * CS * CT * 4);
  int*   hist   = (int*)alloc((size_t)CB * CS * CT * 4);
  float* logZ   = (float*)alloc((size_t)CB * 4);
  float* gold   = (float*)alloc((size_t)CB * 4);
  int*   bar    = (int*)alloc(256);

  auto cvt = [&](const float* src, __bf16* dst, size_t n) {
    f32_to_bf16_kernel<<<(unsigned)((n + 255) / 256), 256, 0, stream>>>(src, dst, (int)n);
  };
  cvt(emb,   emb_bf, (size_t)CV * CE);
  cvt(Wih_f, WihF,   (size_t)CG4 * CE);
  cvt(Whh_f, WhhF,   (size_t)CG4 * CH);
  cvt(Wih_b, WihB,   (size_t)CG4 * CE);
  cvt(Whh_b, WhhB,   (size_t)CG4 * CH);
  wc_pad_kernel<<<64, 256, 0, stream>>>(Wc, Wcp);

  // ------- forward direction
  embed_proj_kernel<<<dim3(1024, 16), 256, 0, stream>>>(ids, emb_bf, WihF,
                                                        bih_f, bhh_f, xproj);
  init_scan_kernel<<<64, 256, 0, stream>>>(hping, bar);
  lstm_scan_kernel<<<32, 256, 0, stream>>>(xproj, WhhF, hping, hsF, bar, 0);

  // ------- backward direction (reuses xproj buffer; stream order protects it)
  embed_proj_kernel<<<dim3(1024, 16), 256, 0, stream>>>(ids, emb_bf, WihB,
                                                        bih_b, bhh_b, xproj);
  init_scan_kernel<<<64, 256, 0, stream>>>(hping, bar);
  lstm_scan_kernel<<<32, 256, 0, stream>>>(xproj, WhhB, hping, hsB, bar, 1);

  classifier_kernel<<<128, 256, 0, stream>>>(hsF, hsB, Wcp, bc, logits);

  crf_logz_kernel<<<CB, 32, 0, stream>>>(logits, start, endv, trans, logZ);
  crf_gold_kernel<<<1, 32, 0, stream>>>(logits, labels, start, endv, trans, gold);
  crf_viterbi_kernel<<<CB, 32, 0, stream>>>(logits, start, endv, trans, hist, out);
  loss_kernel<<<1, 32, 0, stream>>>(gold, logZ, out);
}
